// SpatialAttention_59287728554390
// MI455X (gfx1250) — matrix operand order repaired and checked by reading, not compiled, not run on hardware
//
#include <hip/hip_runtime.h>

// ---------------------------------------------------------------------------
// Spatial attention on gfx1250 (MI455X), wave32 + v_wmma_f32_16x16x32_bf16.
// Layouts chosen so every WMMA fragment is per-lane-contiguous 16B loads:
//   w1b/w2b/wvb : bf16 pre-converted weights (L2-resident)
//   r1t, r2t    : [n][hw][oc]  (bf16, transposed; oc=32 innermost)
//   vb          : [n][c][hw]   (bf16, produced by a flipped GEMM)
//   S = r1^T r2 via one WMMA per 16x16 tile (K = OC = 32)
//   stats: online softmax over j per row i -> m[i], 1/Z[i]
//   attn^T = P^T @ V^T  (P staged transposed in LDS as bf16)
//   out = x + scale*attn  (contiguous fp32 epilogue)
// x tiles are staged once per block through LDS (bf16, transposed, padded),
// shared by all waves -> no per-lane scattered global gathers anywhere.
// bf16 conversion uses native __bf16 fptrunc / convertvector so the backend
// can select hardware cvt (packed) instructions when present.
// ---------------------------------------------------------------------------

typedef __attribute__((ext_vector_type(16))) __bf16 v16bf;
typedef __attribute__((ext_vector_type(2)))  __bf16 v2bf;
typedef __attribute__((ext_vector_type(2)))  float  v2f;
typedef __attribute__((ext_vector_type(8)))  float  v8f;

union Frag16 { v16bf v; unsigned short u[16]; int4 q[2]; };
union Pack8  { int4 q; unsigned w[4]; unsigned short u[8]; };
union Pack4  { uint2 q; unsigned w[2]; unsigned short u[4]; };
union F4     { float4 v; float f[4]; };

constexpr int NB = 4;       // batch
constexpr int C  = 256;     // channels
constexpr int OC = 32;      // reduced channels
constexpr int HW = 4096;    // spatial
constexpr int LP = 40;      // padded LDS row stride (ushorts), keeps 16B align

#define DEVI static __device__ __forceinline__

DEVI unsigned short f2bf(float f) {          // f32 -> bf16 (native fptrunc)
  __bf16 h = (__bf16)f;
  unsigned short u;
  __builtin_memcpy(&u, &h, 2);
  return u;
}

DEVI unsigned cvt2(float lo, float hi) {     // two f32 -> packed bf16x2
  v2f f;  f[0] = lo;  f[1] = hi;
  v2bf h = __builtin_convertvector(f, v2bf); // -> v_cvt_pk_bf16_f32 if present
  unsigned u;
  __builtin_memcpy(&u, &h, 4);
  return u;
}

DEVI int4 ldq(const unsigned short* p) { return *(const int4*)p; }
DEVI float4 ldf4(const float* p) { return *(const float4*)p; }

// Fragment maps (wave32):
//  A elem e (lane l): M = l%16,           K = e + 16*(l/16)
//  B elem e (lane l): K = e + 16*(l/16),  N = l%16
//  D reg  r (lane l): M = r + 8*(l/16),   N = l%16

// ---------------------------------------------------------------------------
// Kernel 0: bulk f32 -> bf16 convert (weights; 4 elems/thread).
// ---------------------------------------------------------------------------
__global__ void __launch_bounds__(256) cvt_bf16_kernel(
    const float* __restrict__ src, unsigned short* __restrict__ dst, int count4)
{
  const int t = (int)(blockIdx.x * blockDim.x + threadIdx.x);
  if (t < count4) {
    F4 f; f.v = ldf4(src + (size_t)t * 4);
    Pack4 p;
    p.w[0] = cvt2(f.f[0], f.f[1]);
    p.w[1] = cvt2(f.f[2], f.f[3]);
    *(uint2*)(dst + (size_t)t * 4) = p.q;
  }
}

// ---------------------------------------------------------------------------
// Kernel 1a: r1t/r2t[n][hw][oc] = (W @ X)^T per batch. One block per
// (n, j-tile); 4 waves = {w1,w2} x {m0=0,16} share one LDS x tile.
// ---------------------------------------------------------------------------
__global__ void __launch_bounds__(128) proj_qk_kernel(
    const unsigned short* __restrict__ W1b, const unsigned short* __restrict__ W2b,
    const float* __restrict__ X,
    unsigned short* __restrict__ r1t, unsigned short* __restrict__ r2t)
{
  __shared__ __align__(16) unsigned short xT[16 * LP];  // [j_local][k_local]
  const int tid  = (int)threadIdx.x;
  const int lane = tid & 31, wave = tid >> 5;           // 4 waves
  const int jt = (int)(blockIdx.x % (HW / 16));
  const int n  = (int)(blockIdx.x / (HW / 16));
  const int j0 = jt * 16;
  const int lh = lane >> 4, ll = lane & 15;
  const float* Xn = X + (size_t)n * C * HW;

  const unsigned short* Wb = (wave < 2) ? W1b : W2b;
  const int m0 = (wave & 1) * 16;

  const int kr = tid >> 2;            // 0..31  (k row of tile)
  const int jc = (tid & 3) * 4;       // 0,4,8,12 (j cols, 4 per thread)

  v8f acc = {};
  for (int k0 = 0; k0 < C; k0 += 32) {
    F4 t; t.v = ldf4(Xn + (size_t)(k0 + kr) * HW + j0 + jc);  // coalesced
    const unsigned p01 = cvt2(t.f[0], t.f[1]);
    const unsigned p23 = cvt2(t.f[2], t.f[3]);
    xT[(jc + 0) * LP + kr] = (unsigned short)p01;
    xT[(jc + 1) * LP + kr] = (unsigned short)(p01 >> 16);
    xT[(jc + 2) * LP + kr] = (unsigned short)p23;
    xT[(jc + 3) * LP + kr] = (unsigned short)(p23 >> 16);
    __syncthreads();

    Frag16 a, b;
    const unsigned short* wrow = Wb + (size_t)(m0 + ll) * C + k0;
    a.q[0] = ldq(wrow + 16 * lh);                // A = W (M=oc, K=cin)
    a.q[1] = ldq(wrow + 16 * lh + 8);
    b.q[0] = ldq(&xT[ll * LP + 16 * lh]);        // B = X (K=cin, N=j)
    b.q[1] = ldq(&xT[ll * LP + 16 * lh + 8]);
    acc = __builtin_amdgcn_wmma_f32_16x16x32_bf16(false, a.v, false, b.v,
                                                  (short)0, acc, false, false);
    __syncthreads();
  }
  Pack8 p;
#pragma unroll
  for (int r = 0; r < 4; ++r) p.w[r] = cvt2(acc[2 * r], acc[2 * r + 1]);
  unsigned short* Ot = ((wave < 2) ? r1t : r2t) + (size_t)n * HW * OC;
  *(int4*)&Ot[(size_t)(j0 + ll) * OC + m0 + 8 * lh] = p.q;
}

// ---------------------------------------------------------------------------
// Kernel 1b: vb[n][c][hw] via flipped GEMM: D(j,c) = X^T @ Wv^T. One block per
// (n, j-tile); 16 waves cover all 16 c-tiles, sharing one LDS x tile.
// ---------------------------------------------------------------------------
__global__ void __launch_bounds__(512) proj_v_kernel(
    const unsigned short* __restrict__ Wvb, const float* __restrict__ X,
    unsigned short* __restrict__ vb)
{
  __shared__ __align__(16) unsigned short xT[16 * LP];  // [j_local][k_local]
  const int tid  = (int)threadIdx.x;
  const int lane = tid & 31, wave = tid >> 5;           // 16 waves
  const int jt = (int)(blockIdx.x % (HW / 16));
  const int n  = (int)(blockIdx.x / (HW / 16));
  const int j0 = jt * 16;
  const int lh = lane >> 4, ll = lane & 15;
  const float* Xn = X + (size_t)n * C * HW;
  const int c0 = wave * 16;

  const int kr = tid >> 4;            // 0..31
  const int jc = tid & 15;            // 0..15

  v8f acc = {};
  for (int k0 = 0; k0 < C; k0 += 32) {
    xT[jc * LP + kr] = f2bf(Xn[(size_t)(k0 + kr) * HW + j0 + jc]);  // coalesced
    __syncthreads();

    Frag16 a, b;
    a.q[0] = ldq(&xT[ll * LP + 16 * lh]);        // A = X^T (M=j, K=cin)
    a.q[1] = ldq(&xT[ll * LP + 16 * lh + 8]);
    const unsigned short* brow = Wvb + (size_t)(c0 + ll) * C + k0 + 16 * lh;
    b.q[0] = ldq(brow);                          // B = Wv^T (K=cin, N=cout)
    b.q[1] = ldq(brow + 8);
    acc = __builtin_amdgcn_wmma_f32_16x16x32_bf16(false, a.v, false, b.v,
                                                  (short)0, acc, false, false);
    __syncthreads();
  }
  Pack8 p;
#pragma unroll
  for (int r = 0; r < 4; ++r) p.w[r] = cvt2(acc[2 * r], acc[2 * r + 1]);
  unsigned short* Vn = vb + (size_t)n * C * HW;
  *(int4*)&Vn[(size_t)(c0 + ll) * HW + j0 + 8 * lh] = p.q;
}

// ---------------------------------------------------------------------------
// Kernel 2: online softmax stats of S[i,:]. One wave per (n, 16-row i-tile).
// A-frag (r1^T) loaded once (2x b128); per j-tile: 2x b128 + 1 WMMA.
// ---------------------------------------------------------------------------
__global__ void __launch_bounds__(128) stats_kernel(
    const unsigned short* __restrict__ r1t, const unsigned short* __restrict__ r2t,
    float* __restrict__ mrow, float* __restrict__ rzrow)
{
  const int lane = threadIdx.x & 31;
  const int wid  = (int)((blockIdx.x * blockDim.x + threadIdx.x) >> 5);
  const int it = wid % (HW / 16);
  const int n  = wid / (HW / 16);
  const int i0 = it * 16;
  const int lh = (lane >> 4), ll = (lane & 15);
  const unsigned short* R1 = r1t + (size_t)n * HW * OC;
  const unsigned short* R2 = r2t + (size_t)n * HW * OC;

  Frag16 a;
  const unsigned short* arow = R1 + (size_t)(i0 + ll) * OC;
  a.q[0] = ldq(arow + 16 * lh);
  a.q[1] = ldq(arow + 16 * lh + 8);

  float m[8], s[8];
#pragma unroll
  for (int r = 0; r < 8; ++r) { m[r] = -3.0e38f; s[r] = 0.0f; }

  for (int jt = 0; jt < HW / 16; ++jt) {
    const int j0 = jt * 16;
    const unsigned short* brow = R2 + (size_t)(j0 + ll) * OC + (lh << 4);
    Frag16 b;
    b.q[0] = ldq(brow);
    b.q[1] = ldq(brow + 8);
    __builtin_prefetch(brow + 16 * OC, 0, 1);           // next j-tile stream
    v8f S = {};
    S = __builtin_amdgcn_wmma_f32_16x16x32_bf16(false, a.v, false, b.v,
                                                (short)0, S, false, false);
#pragma unroll
    for (int r = 0; r < 8; ++r) {
      float v = S[r];
      float t = v;                                      // row max (16 lanes)
      t = fmaxf(t, __shfl_xor(t, 1, 32));
      t = fmaxf(t, __shfl_xor(t, 2, 32));
      t = fmaxf(t, __shfl_xor(t, 4, 32));
      t = fmaxf(t, __shfl_xor(t, 8, 32));
      const float mn = fmaxf(m[r], t);
      float p = __expf(v - mn);                         // row sum of exp
      p += __shfl_xor(p, 1, 32);
      p += __shfl_xor(p, 2, 32);
      p += __shfl_xor(p, 4, 32);
      p += __shfl_xor(p, 8, 32);
      s[r] = s[r] * __expf(m[r] - mn) + p;
      m[r] = mn;
    }
  }
  if (ll == 0) {                                        // lanes 0 and 16
    float* Mn = mrow  + (size_t)n * HW;
    float* Zn = rzrow + (size_t)n * HW;
#pragma unroll
    for (int r = 0; r < 8; ++r) {
      const int i = i0 + r + (lh << 3);
      Mn[i] = m[r];
      Zn[i] = 1.0f / s[r];
    }
  }
}

// ---------------------------------------------------------------------------
// Kernel 3: attn^T tiles: D(j,c) = P^T @ V^T; out = x + scale*attn.
// 8 waves/block, one (n, 16-col j-tile) per block. Per 32-i chunk:
//   waves 0,1: score tiles (1 WMMA each), exp*rcpZ, P^T -> LDS (1 b128/lane);
//   all 8 waves: A = P^T (2x ds b128), B = v rows (2x b128), 2 WMMAs.
// Epilogue: per-lane contiguous fp32 rows -> float4 residual fuse.
// ---------------------------------------------------------------------------
__global__ void __launch_bounds__(256) attn_kernel(
    const float* __restrict__ X, const unsigned short* __restrict__ vb,
    const unsigned short* __restrict__ r1t, const unsigned short* __restrict__ r2t,
    const float* __restrict__ mrow, const float* __restrict__ rzrow,
    const float* __restrict__ scale, float* __restrict__ out)
{
  __shared__ __align__(16) unsigned short PtT[16 * 32];  // [j_local][i_local]
  const int lane = threadIdx.x & 31;
  const int wave = (int)(threadIdx.x >> 5);             // 0..7
  const int jt = (int)(blockIdx.x % (HW / 16));
  const int n  = (int)(blockIdx.x / (HW / 16));
  const int j0 = jt * 16;
  const int lh = (lane >> 4), ll = (lane & 15);
  const unsigned short* R1 = r1t + (size_t)n * HW * OC;
  const unsigned short* R2 = r2t + (size_t)n * HW * OC;
  const unsigned short* Vn = vb  + (size_t)n * C * HW;
  const float* Mn = mrow  + (size_t)n * HW;
  const float* Zn = rzrow + (size_t)n * HW;

  Frag16 bS;                              // B frag of score WMMA (fixed j-tile)
  if (wave < 2) {
    const unsigned short* brow = R2 + (size_t)(j0 + ll) * OC + (lh << 4);
    bS.q[0] = ldq(brow);
    bS.q[1] = ldq(brow + 8);
  }

  const int c0 = wave * 32;               // 32 channel cols per wave
  v8f acc0 = {}, acc1 = {};

  for (int i0 = 0; i0 < HW; i0 += 32) {
    if (wave < 2) {
      const int ib = i0 + wave * 16;
      const unsigned short* arow = R1 + (size_t)(ib + ll) * OC;
      Frag16 aS;
      aS.q[0] = ldq(arow + 16 * lh);
      aS.q[1] = ldq(arow + 16 * lh + 8);
      v8f S = {};
      S = __builtin_amdgcn_wmma_f32_16x16x32_bf16(false, aS.v, false, bS.v,
                                                  (short)0, S, false, false);
      float pf[8];
#pragma unroll
      for (int r = 0; r < 8; ++r) {
        const int ig = i0 + wave * 16 + r + (lh << 3);
        pf[r] = __expf(S[r] - Mn[ig]) * Zn[ig];
      }
      Pack8 pk;
#pragma unroll
      for (int r = 0; r < 4; ++r) pk.w[r] = cvt2(pf[2 * r], pf[2 * r + 1]);
      // S-frag lane (row i=r+8lh, col j=ll) -> PtT[ll][16*wave + 8*lh + r]
      *(int4*)&PtT[ll * 32 + wave * 16 + 8 * lh] = pk.q;
    }
    __syncthreads();

    Frag16 aP, b0, b1;
    aP.q[0] = ldq(&PtT[ll * 32 + 16 * lh]);             // A = P^T (M=j, K=i)
    aP.q[1] = ldq(&PtT[ll * 32 + 16 * lh + 8]);
    const unsigned short* v0row = Vn + (size_t)(c0 + ll) * HW + i0 + (lh << 4);
    const unsigned short* v1row = v0row + (size_t)16 * HW;
    b0.q[0] = ldq(v0row);  b0.q[1] = ldq(v0row + 8);    // B = V^T (K=i, N=c)
    b1.q[0] = ldq(v1row);  b1.q[1] = ldq(v1row + 8);
    __builtin_prefetch(v0row + 32, 0, 1);               // next i-chunk stream
    __builtin_prefetch(v1row + 32, 0, 1);
    acc0 = __builtin_amdgcn_wmma_f32_16x16x32_bf16(false, aP.v, false, b0.v,
                                                   (short)0, acc0, false, false);
    acc1 = __builtin_amdgcn_wmma_f32_16x16x32_bf16(false, aP.v, false, b1.v,
                                                   (short)0, acc1, false, false);
    __syncthreads();
  }

  // D reg r: j = j0 + r + 8*lh (contiguous in r), c = c0 (+16) + ll
  const float sc = scale[0];
  const float* Xn = X   + (size_t)n * C * HW;
  float*       On = out + (size_t)n * C * HW;
#pragma unroll
  for (int t = 0; t < 2; ++t) {
    const v8f& acc = t ? acc1 : acc0;
    const size_t base = (size_t)(c0 + 16 * t + ll) * HW + j0 + 8 * lh;
    F4 xa, xb, oa, ob;
    xa.v = ldf4(Xn + base);
    xb.v = ldf4(Xn + base + 4);
#pragma unroll
    for (int r = 0; r < 4; ++r) {
      oa.f[r] = xa.f[r] + sc * acc[r];
      ob.f[r] = xb.f[r] + sc * acc[4 + r];
    }
    *(float4*)(On + base)     = oa.v;
    *(float4*)(On + base + 4) = ob.v;
  }
}

// ---------------------------------------------------------------------------
extern "C" void kernel_launch(void* const* d_in, const int* in_sizes, int n_in,
                              void* d_out, int out_size, void* d_ws, size_t ws_size,
                              hipStream_t stream)
{
  const float* x     = (const float*)d_in[0];
  const float* w1    = (const float*)d_in[1];
  const float* w2    = (const float*)d_in[2];
  const float* wv    = (const float*)d_in[3];
  const float* scale = (const float*)d_in[4];
  float* out = (float*)d_out;

  // Workspace carve-up (~10.3 MB total)
  unsigned short* r1t = (unsigned short*)d_ws;                  // 1 MB
  unsigned short* r2t = r1t + (size_t)NB * HW * OC;             // 1 MB
  unsigned short* vb  = r2t + (size_t)NB * HW * OC;             // 8 MB
  float* mrow = (float*)(vb + (size_t)NB * C * HW);             // 64 KB
  float* rz   = mrow + (size_t)NB * HW;                         // 64 KB
  unsigned short* w1b = (unsigned short*)(rz + (size_t)NB * HW);// 16 KB
  unsigned short* w2b = w1b + (size_t)OC * C;                   // 16 KB
  unsigned short* wvb = w2b + (size_t)OC * C;                   // 128 KB

  // Pre-convert weights to bf16 (L2-resident, read by every proj block).
  cvt_bf16_kernel<<<(OC * C / 4 + 255) / 256, 256, 0, stream>>>(w1, w1b, OC * C / 4);
  cvt_bf16_kernel<<<(OC * C / 4 + 255) / 256, 256, 0, stream>>>(w2, w2b, OC * C / 4);
  cvt_bf16_kernel<<<(C * C / 4 + 255) / 256, 256, 0, stream>>>(wv, wvb, C * C / 4);

  proj_qk_kernel<<<NB * (HW / 16), 128, 0, stream>>>(w1b, w2b, x, r1t, r2t);
  proj_v_kernel<<<NB * (HW / 16), 512, 0, stream>>>(wvb, x, vb);
  stats_kernel<<<NB * (HW / 16) / 4, 128, 0, stream>>>(r1t, r2t, mrow, rz);
  attn_kernel<<<NB * (HW / 16), 256, 0, stream>>>(x, vb, r1t, r2t, mrow, rz,
                                                  scale, out);
}
